// FFN_53420803228395
// MI455X (gfx1250) — matrix operand order repaired and checked by reading, not compiled, not run on hardware
//
#include <hip/hip_runtime.h>

// ---------------------------------------------------------------------------
// Fully-fused MLP  2 -> 1024 -> 512 -> 256 -> 3  (ReLU between layers)
// CDNA5 / gfx1250: wave32, v_wmma_f32_16x16x32_f16, 320KB LDS per WGP.
//
// Strategy:
//   * prep kernel: transpose+convert W1/W2/W3 (fp32 row-major [Fin][Fout])
//     into f16 W^T row-major [Fout][Fin] in d_ws (W3 zero-padded 3->16 rows).
//   * fused kernel: 256 threads (8 waves) per block, 64 batch rows per block.
//     Activations ping-pong between two padded f16 LDS buffers; weights are
//     streamed from L2 once per block; all GEMMs via WMMA f16 -> f32 acc.
// ---------------------------------------------------------------------------

typedef __attribute__((ext_vector_type(16))) _Float16 v16h;
typedef __attribute__((ext_vector_type(8)))  _Float16 v8h;
typedef __attribute__((ext_vector_type(8)))  float    v8f;

#define M_TILE   64
#define THREADS  256
#define NWAVES   8

// LDS activation row strides (halves), padded +8 so that the 16 lanes of an
// A-fragment ds_load_b128 walk 4-dword-stride across all 64 banks.
#define S0 1032   // layer0 output, 1024 wide
#define S1 520    // layer1 output,  512 wide
#define S2 264    // layer2 output,  256 wide

#define OFF_BUF0 0
#define OFF_BUF1 (M_TILE * S0)                 // 66048 halves
#define OFF_CS   (OFF_BUF1 + M_TILE * S1)      // 99328 halves (coord cache, f32)
#define SMEM_HALVES (OFF_CS + 2 * M_TILE * 2)  // +128 floats
#define SMEM_BYTES  (SMEM_HALVES * 2)          // 199168 B  (< 320KB WGP LDS)

// d_ws layout (halves)
#define WT1_OFF 0
#define WT1_SZ  (512 * 1024)
#define WT2_OFF (WT1_OFF + WT1_SZ)
#define WT2_SZ  (256 * 512)
#define WT3_OFF (WT2_OFF + WT2_SZ)
#define WT3_SZ  (16 * 256)
#define WT_TOTAL (WT3_OFF + WT3_SZ)            // 659456 halves = ~1.26 MB

// ---------------------------------------------------------------------------
// Weight transpose + fp32 -> fp16 convert (one-time tiny pass, L2 resident)
// ---------------------------------------------------------------------------
__global__ void prep_weights(const float* __restrict__ W1,
                             const float* __restrict__ W2,
                             const float* __restrict__ W3,
                             _Float16* __restrict__ WT) {
  int i = blockIdx.x * THREADS + threadIdx.x;
  float v;
  if (i < WT1_SZ) {                      // WT1: [512][1024] <- W1 [1024][512]
    int fo = i >> 10, fi = i & 1023;
    v = W1[fi * 512 + fo];
  } else if (i < WT2_OFF + WT2_SZ) {     // WT2: [256][512]  <- W2 [512][256]
    int j = i - WT2_OFF;
    int fo = j >> 9, fi = j & 511;
    v = W2[fi * 256 + fo];
  } else if (i < WT_TOTAL) {             // WT3: [16][256]   <- W3 [256][3], pad
    int j = i - WT3_OFF;
    int fo = j >> 8, fi = j & 255;
    v = (fo < 3) ? W3[fi * 3 + fo] : 0.0f;
  } else {
    return;
  }
  WT[i] = (_Float16)v;
}

// ---------------------------------------------------------------------------
// One hidden GEMM layer: out[64 x 16*ntiles] = relu(in[64 x Fin] @ WT^T + b)
// Each wave owns column-tiles nt (strided); one B fragment is reused across
// the 4 row-tiles (4 independent accumulator chains -> no WMMA RAW stalls,
// and every weight byte is read exactly once per block).
// ---------------------------------------------------------------------------
__device__ __forceinline__ void relu_store(v8f acc, _Float16* op, int sOut) {
#pragma unroll
  for (int j = 0; j < 8; ++j) {
    float v = acc[j];
    v = v > 0.0f ? v : 0.0f;
    op[j * sOut] = (_Float16)v;
  }
}

__device__ __forceinline__ void wmma_layer(const _Float16* __restrict__ WT,
                                           const float* __restrict__ bias,
                                           const _Float16* in, int sIn, int Fin,
                                           _Float16* out, int sOut, int ntiles,
                                           int wave, int lane) {
  const int nlo  = lane & 15;
  const int hi8  = (lane & 16) ? 8 : 0;   // C row-offset
  const int hi16 = hi8 * 2;               // B K-offset

  for (int nt = wave; nt < ntiles; nt += NWAVES) {
    const int col = nt * 16 + nlo;
    const float bv = bias[col];
    v8f acc0 = {bv, bv, bv, bv, bv, bv, bv, bv};
    v8f acc1 = acc0, acc2 = acc0, acc3 = acc0;

    const _Float16* bptr = WT + (size_t)col * Fin + hi16;
    const _Float16* aptr = in + nlo * sIn + hi16;

    for (int k0 = 0; k0 < Fin; k0 += 32) {
      v16h b = *(const v16h*)(bptr + k0);   // 32B contiguous: 2x global_load_b128
#pragma unroll
      for (int mt = 0; mt < 4; ++mt) {
        const _Float16* ar = aptr + mt * 16 * sIn + k0;
        v8h a0 = *(const v8h*)(ar);         // ds_load_b128
        v8h a1 = *(const v8h*)(ar + 8);     // ds_load_b128
        v16h a = __builtin_shufflevector(a0, a1, 0, 1, 2, 3, 4, 5, 6, 7,
                                         8, 9, 10, 11, 12, 13, 14, 15);
        v8f& acc = (mt == 0) ? acc0 : (mt == 1) ? acc1 : (mt == 2) ? acc2 : acc3;
        acc = __builtin_amdgcn_wmma_f32_16x16x32_f16(
            false, a, false, b, (short)0, acc, false, false);
      }
    }
    _Float16* op = out + hi8 * sOut + col;
    relu_store(acc0, op + 0 * 16 * sOut, sOut);
    relu_store(acc1, op + 1 * 16 * sOut, sOut);
    relu_store(acc2, op + 2 * 16 * sOut, sOut);
    relu_store(acc3, op + 3 * 16 * sOut, sOut);
  }
}

// ---------------------------------------------------------------------------
// Fused MLP kernel: one block = 64 rows of the batch.
// ---------------------------------------------------------------------------
__global__ __launch_bounds__(THREADS) void fused_mlp(
    const float* __restrict__ coord,
    const float* __restrict__ W0, const float* __restrict__ b0,
    const _Float16* __restrict__ WT1, const float* __restrict__ b1,
    const _Float16* __restrict__ WT2, const float* __restrict__ b2,
    const _Float16* __restrict__ WT3, const float* __restrict__ b3,
    float* __restrict__ out) {
  extern __shared__ _Float16 smem[];
  _Float16* buf0 = smem + OFF_BUF0;
  _Float16* buf1 = smem + OFF_BUF1;
  float* cs = (float*)(smem + OFF_CS);

  const int t = threadIdx.x;
  const int wave = t >> 5;
  const int lane = t & 31;
  const int blockRow = blockIdx.x * M_TILE;

  // stage the 64x2 input coords
  if (t < 2 * M_TILE) cs[t] = coord[blockRow * 2 + t];
  __syncthreads();

  // ---- layer 0: 2 -> 1024 (VALU; WMMA would waste 30/32 of K) -------------
  for (int idx = t; idx < M_TILE * 1024; idx += THREADS) {
    const int r = idx >> 10;
    const int n = idx & 1023;
    float v = fmaf(cs[2 * r], W0[n], fmaf(cs[2 * r + 1], W0[1024 + n], b0[n]));
    v = v > 0.0f ? v : 0.0f;
    buf0[r * S0 + n] = (_Float16)v;
  }
  __syncthreads();

  // ---- layer 1: 1024 -> 512 (32 n-tiles over 8 waves) ---------------------
  wmma_layer(WT1, b1, buf0, S0, 1024, buf1, S1, 32, wave, lane);
  __syncthreads();

  // ---- layer 2: 512 -> 256 (16 n-tiles) -----------------------------------
  wmma_layer(WT2, b2, buf1, S1, 512, buf0, S2, 16, wave, lane);
  __syncthreads();

  // ---- layer 3: 256 -> 3 (padded to 16 cols; one row-tile per wave 0..3) --
  if (wave < 4) {
    const int nlo = lane & 15;
    const int hi8 = (lane & 16) ? 8 : 0;
    const int mt = wave;
    const float bv = (nlo < 3) ? b3[nlo] : 0.0f;
    v8f acc = {bv, bv, bv, bv, bv, bv, bv, bv};
    const _Float16* bptr = WT3 + nlo * 256 + hi8 * 2;
    const _Float16* aptr = buf0 + (mt * 16 + nlo) * S2 + hi8 * 2;
#pragma unroll
    for (int k0 = 0; k0 < 256; k0 += 32) {
      v16h b = *(const v16h*)(bptr + k0);
      v8h a0 = *(const v8h*)(aptr + k0);
      v8h a1 = *(const v8h*)(aptr + k0 + 8);
      v16h a = __builtin_shufflevector(a0, a1, 0, 1, 2, 3, 4, 5, 6, 7,
                                       8, 9, 10, 11, 12, 13, 14, 15);
      acc = __builtin_amdgcn_wmma_f32_16x16x32_f16(
          false, a, false, b, (short)0, acc, false, false);
    }
    if (nlo < 3) {
      const int r0 = blockRow + mt * 16 + hi8;
#pragma unroll
      for (int j = 0; j < 8; ++j) out[(size_t)(r0 + j) * 3 + nlo] = acc[j];
    }
  }
}

// ---------------------------------------------------------------------------
extern "C" void kernel_launch(void* const* d_in, const int* in_sizes, int n_in,
                              void* d_out, int out_size, void* d_ws, size_t ws_size,
                              hipStream_t stream) {
  const float* coord = (const float*)d_in[0];
  const float* W0 = (const float*)d_in[1];
  const float* b0 = (const float*)d_in[2];
  const float* W1 = (const float*)d_in[3];
  const float* b1 = (const float*)d_in[4];
  const float* W2 = (const float*)d_in[5];
  const float* b2 = (const float*)d_in[6];
  const float* W3 = (const float*)d_in[7];
  const float* b3 = (const float*)d_in[8];

  _Float16* WT = (_Float16*)d_ws;  // needs WT_TOTAL*2 ~ 1.26 MB of scratch

  const int N = in_sizes[0] / 2;           // 262144
  const int prep_blocks = (WT_TOTAL + THREADS - 1) / THREADS;

  prep_weights<<<prep_blocks, THREADS, 0, stream>>>(W1, W2, W3, WT);
  fused_mlp<<<N / M_TILE, THREADS, SMEM_BYTES, stream>>>(
      coord, W0, b0,
      WT + WT1_OFF, b1,
      WT + WT2_OFF, b2,
      WT + WT3_OFF, b3,
      (float*)d_out);
}
